// HighwayLayer_5798205850096
// MI455X (gfx1250) — matrix operand order repaired and checked by reading, not compiled, not run on hardware
//
#include <hip/hip_runtime.h>
#include <hip/hip_bf16.h>

typedef __attribute__((ext_vector_type(16))) __bf16 v16bf;
typedef __attribute__((ext_vector_type(8)))  __bf16 v8bf;
typedef __attribute__((ext_vector_type(4)))  __bf16 v4bf;
typedef __attribute__((ext_vector_type(8)))  float  v8f;

#define D        512
#define SPLIT    128
#define BM       64
#define BN       128
#define KBLK     128
#define APAD     8      // LDS row pad (bf16 elems) -> row stride 272B, breaks bank conflicts

__device__ __forceinline__ __bf16 f2bf(float f) {
    union { float f; unsigned u; } v; v.f = f;
    unsigned r = v.u + 0x7fffu + ((v.u >> 16) & 1u);   // round-to-nearest-even
    unsigned short h = (unsigned short)(r >> 16);
    return __builtin_bit_cast(__bf16, h);
}

__device__ __forceinline__ v16bf combine16(v8bf lo, v8bf hi) {
    v16bf r;
#pragma unroll
    for (int e = 0; e < 8; ++e) { r[e] = lo[e]; r[8 + e] = hi[e]; }
    return r;
}

// tanh via hardware v_exp_f32: tanh(x) = sign(x) * (1 - e) / (1 + e), e = exp(-2|x|)
__device__ __forceinline__ float fast_tanh(float x) {
    float ax = fabsf(x);
    float e  = __expf(-2.0f * ax);
    float t  = (1.0f - e) / (1.0f + e);
    return copysignf(t, x);
}

// ---------------- Kernel 1: column-wise partial sums over row chunks ----------------
__global__ void hw_stats_partial(const float* __restrict__ x1, const float* __restrict__ x2,
                                 float* __restrict__ part, int N) {
    const int col   = blockIdx.x * 256 + threadIdx.x;         // 0..511
    const int chunk = (N + SPLIT - 1) / SPLIT;
    const int r0    = blockIdx.y * chunk;
    const int r1    = min(N, r0 + chunk);
    float s1 = 0.f, q1 = 0.f, s2 = 0.f, q2 = 0.f;
    for (int r = r0; r < r1; ++r) {
        float a = x1[(size_t)r * D + col];
        float b = x2[(size_t)r * D + col];
        s1 += a; q1 += a * a;
        s2 += b; q2 += b * b;
    }
    const int o = blockIdx.y * D + col;
    part[o]                 = s1;
    part[SPLIT * D + o]     = q1;
    part[2 * SPLIT * D + o] = s2;
    part[3 * SPLIT * D + o] = q2;
}

// ---------------- Kernel 2: finalize mean / rstd ----------------
__global__ void hw_stats_final(const float* __restrict__ part, float* __restrict__ stats, int N) {
    const int col = blockIdx.x * 256 + threadIdx.x;           // 0..511
    float s1 = 0.f, q1 = 0.f, s2 = 0.f, q2 = 0.f;
    for (int i = 0; i < SPLIT; ++i) {
        s1 += part[i * D + col];
        q1 += part[SPLIT * D + i * D + col];
        s2 += part[2 * SPLIT * D + i * D + col];
        q2 += part[3 * SPLIT * D + i * D + col];
    }
    const float inv = 1.0f / (float)N;
    float m1 = s1 * inv, m2 = s2 * inv;
    float v1 = q1 * inv - m1 * m1;      // biased var (jnp.var ddof=0)
    float v2 = q2 * inv - m2 * m2;
    stats[col]         = m1;
    stats[D + col]     = __frsqrt_rn(v1 + 1e-3f);
    stats[2 * D + col] = m2;
    stats[3 * D + col] = __frsqrt_rn(v2 + 1e-3f);
}

// ---------------- Kernel 3: W[k][n] -> Wt[n][k], fp32 -> bf16 ----------------
__global__ void hw_transpose_bf16(const float* __restrict__ W, __bf16* __restrict__ Wt) {
    __shared__ float tile[32][33];
    const int tx = threadIdx.x, ty = threadIdx.y;             // 32 x 8
    const int kb = blockIdx.y * 32, nb = blockIdx.x * 32;
#pragma unroll
    for (int i = 0; i < 4; ++i)
        tile[ty + i * 8][tx] = W[(size_t)(kb + ty + i * 8) * D + nb + tx];
    __syncthreads();
#pragma unroll
    for (int i = 0; i < 4; ++i)
        Wt[(size_t)(nb + ty + i * 8) * D + kb + tx] = f2bf(tile[tx][ty + i * 8]);
}

// ---------------- Kernel 4: fused BN + bf16 WMMA GEMM + gated tanh ----------------
__global__ __launch_bounds__(256)
void hw_main(const float* __restrict__ x1, const float* __restrict__ x2,
             const __bf16* __restrict__ Wt, const float* __restrict__ stats,
             float* __restrict__ out, int N) {
    __shared__ __bf16 As[BM][KBLK + APAD];   // normalized x1 tile (bf16)
    __shared__ __bf16 Bs[BN][KBLK + APAD];   // Wt panel: Bs[n][k]

    const float* mean1 = stats;
    const float* rstd1 = stats + D;
    const float* mean2 = stats + 2 * D;
    const float* rstd2 = stats + 3 * D;

    // column tile on x (fast-varying) so the 4 column tiles sharing an x1 row
    // panel run back-to-back -> x1 panel stays hot in the 192MB L2.
    const int n0   = blockIdx.x * BN;
    const int m0   = blockIdx.y * BM;
    const int tid  = threadIdx.x;
    const int lane = tid & 31;
    const int wave = tid >> 5;
    const int wr   = wave >> 2;      // 0..1  (row group of 32)
    const int wc   = wave & 3;       // 0..3  (col group of 32)
    const int half = lane >> 4;      // K-half selector
    const int l16  = lane & 15;

    v8f acc[2][2] = {};

    for (int kc = 0; kc < D / KBLK; ++kc) {
        const int k0 = kc * KBLK;

        // ---- B panel: async DMA global -> LDS (ASYNCcnt), overlapped with A normalize
        // 128x128 bf16 = 2048 x 16B, 8 per thread
#pragma unroll
        for (int it = 0; it < 8; ++it) {
            int idx = tid + it * 256;
            int row = idx >> 4;          // 16 x 16B per row
            int c   = idx & 15;
            unsigned lds_off = (unsigned)(size_t)&Bs[row][c * 8];     // LDS aperture: addr[31:0]
            const __bf16* g  = Wt + (size_t)(n0 + row) * D + k0 + c * 8;
            asm volatile("global_load_async_to_lds_b128 %0, %1, off"
                         :: "v"(lds_off), "v"(g) : "memory");
        }

        // ---- A tile: load + normalize 64x128 fp32 -> bf16 (2048 float4, 8 per thread)
#pragma unroll
        for (int it = 0; it < 8; ++it) {
            int idx = tid + it * 256;
            int r   = idx >> 5;          // 32 float4 per row
            int c4  = idx & 31;
            int m   = m0 + r;
            int k   = k0 + c4 * 4;
            float4 x = make_float4(0.f, 0.f, 0.f, 0.f);
            if (m < N) x = *(const float4*)(x1 + (size_t)m * D + k);
            v4bf o;
            o[0] = f2bf((x.x - mean1[k + 0]) * rstd1[k + 0]);
            o[1] = f2bf((x.y - mean1[k + 1]) * rstd1[k + 1]);
            o[2] = f2bf((x.z - mean1[k + 2]) * rstd1[k + 2]);
            o[3] = f2bf((x.w - mean1[k + 3]) * rstd1[k + 3]);
            *(v4bf*)(&As[r][c4 * 4]) = o;
        }

        asm volatile("s_wait_asynccnt 0x0" ::: "memory");   // B DMA complete
        __syncthreads();

        // ---- WMMA over this K chunk
#pragma unroll
        for (int kk = 0; kk < KBLK; kk += 32) {
            v16bf a[2], b[2];
#pragma unroll
            for (int i = 0; i < 2; ++i) {
                // A layout (ISA 7.12.2): elems 0..7 -> K = half*8 + 0..7 ; 8..15 -> K = 16 + half*8 + 0..7
                const __bf16* ap = &As[wr * 32 + i * 16 + l16][kk + half * 8];
                a[i] = combine16(*(const v8bf*)ap, *(const v8bf*)(ap + 16));
            }
#pragma unroll
            for (int j = 0; j < 2; ++j) {
                // B layout: lane's col = l16, elems 0..7 -> K = half*8 + 0..7 ; 8..15 -> K = 16 + half*8 + 0..7
                const __bf16* bp = &Bs[wc * 32 + j * 16 + l16][kk + half * 8];
                b[j] = combine16(*(const v8bf*)bp, *(const v8bf*)(bp + 16));
            }
#pragma unroll
            for (int i = 0; i < 2; ++i)
#pragma unroll
                for (int j = 0; j < 2; ++j)
                    acc[i][j] = __builtin_amdgcn_wmma_f32_16x16x32_bf16(
                        false, a[i], false, b[j], (short)0, acc[i][j], false, false);
        }
        __syncthreads();
    }

    // ---- epilogue: gate = relu(tanh(acc)) = tanh(max(acc,0)); out = tanh(b2 + g*(b1-b2))
#pragma unroll
    for (int i = 0; i < 2; ++i) {
#pragma unroll
        for (int j = 0; j < 2; ++j) {
            const int mbase = m0 + wr * 32 + i * 16 + half * 8;   // C: M = v + 8*half
            const int nn    = n0 + wc * 32 + j * 16 + l16;        // C: N = l16
            const float mu1 = mean1[nn], r1 = rstd1[nn];
            const float mu2 = mean2[nn], r2 = rstd2[nn];
#pragma unroll
            for (int v = 0; v < 8; ++v) {
                int m = mbase + v;
                if (m < N) {
                    float g  = fast_tanh(fmaxf(acc[i][j][v], 0.f));
                    float b1 = (x1[(size_t)m * D + nn] - mu1) * r1;
                    float b2 = (x2[(size_t)m * D + nn] - mu2) * r2;
                    out[(size_t)m * D + nn] = fast_tanh(b2 + g * (b1 - b2));
                }
            }
        }
    }
}

extern "C" void kernel_launch(void* const* d_in, const int* in_sizes, int n_in,
                              void* d_out, int out_size, void* d_ws, size_t ws_size,
                              hipStream_t stream) {
    const float* x1 = (const float*)d_in[0];
    const float* x2 = (const float*)d_in[1];
    const float* W  = (const float*)d_in[2];
    const int N = in_sizes[0] / D;           // 100000

    float*  part  = (float*)d_ws;                        // 4*SPLIT*D floats
    float*  stats = part + 4 * SPLIT * D;                // 4*D floats
    __bf16* Wt    = (__bf16*)(stats + 4 * D);            // D*D bf16

    hw_stats_partial<<<dim3(D / 256, SPLIT), 256, 0, stream>>>(x1, x2, part, N);
    hw_stats_final<<<D / 256, 256, 0, stream>>>(part, stats, N);
    hw_transpose_bf16<<<dim3(D / 32, D / 32), dim3(32, 8), 0, stream>>>(W, Wt);
    hw_main<<<dim3(D / BN, (N + BM - 1) / BM), 256, 0, stream>>>(x1, x2, Wt, stats,
                                                                 (float*)d_out, N);
}
